// TextExperts_6614249635977
// MI455X (gfx1250) — matrix operand order repaired and checked by reading, not compiled, not run on hardware
//
#include <hip/hip_runtime.h>

// MoE expert FFN for MI455X (gfx1250, wave32, WMMA, TDM).
// GEMM1 (x @ W_gate_up) fused with SwiGLU -> bf16 act in d_ws
// GEMM2 (act @ W_down, A-tile staged via Tensor Data Mover) -> fp32 out
//
// E=8 experts, T=1024 tokens/expert, H=2048 hidden, F=5632 expert dim.

#define NE 8
#define NT 1024
#define NH 2048
#define NF 5632
#define N2F 11264

#define LDA 72   // padded LDS stride (elements) for 64-deep K tiles
#define LDB 72

typedef __attribute__((ext_vector_type(2)))  __bf16   v2bf;
typedef __attribute__((ext_vector_type(16))) __bf16   v16bf;
typedef __attribute__((ext_vector_type(8)))  float    v8f;
typedef __attribute__((ext_vector_type(4)))  unsigned u32x4;
typedef __attribute__((ext_vector_type(8)))  int      i32x8;
typedef __attribute__((ext_vector_type(4)))  int      i32x4;

#if __has_builtin(__builtin_amdgcn_tensor_load_to_lds) && __has_builtin(__builtin_amdgcn_s_wait_tensorcnt)
#define USE_TDM 1
#else
#define USE_TDM 0
#endif

struct U32x8 { uint4 lo, hi; };

// native fp32 -> bf16 (backend emits hw cvt on gfx1250)
__device__ __forceinline__ unsigned pk2(float lo, float hi) {
    v2bf p; p.x = (__bf16)lo; p.y = (__bf16)hi;
    return __builtin_bit_cast(unsigned, p);
}

// A fragment, 16x32 bf16 (ISA 7.12.2): lane<16: row=lane,  K = {k0..k0+7, k0+16..k0+23}
//                                      lane>=16: row=lane-16, K += 8 / 24
__device__ __forceinline__ v16bf ld_frag_a(const __bf16* lds, int rb, int k0, int lane) {
    const __bf16* p = lds + (rb + (lane & 15)) * LDA + k0 + ((lane >> 4) << 3);
    U32x8 t;
    t.lo = *(const uint4*)(p);        // K .. K+7   -> VGPR0..3
    t.hi = *(const uint4*)(p + 16);   // K+16..K+23 -> VGPR4..7
    return __builtin_bit_cast(v16bf, t);
}

// B fragment, 32x16 bf16 from transposed LDS tile [N][K]:
// lane<16: col=lane, K = {k0..k0+7, k0+16..k0+23} ; lane>=16: col=lane-16, K += 8 / 24
__device__ __forceinline__ v16bf ld_frag_b(const __bf16* lds, int nb, int k0, int lane) {
    const __bf16* p = lds + (nb + (lane & 15)) * LDB + k0 + ((lane >> 4) << 3);
    U32x8 t;
    t.lo = *(const uint4*)(p);        // K .. K+7
    t.hi = *(const uint4*)(p + 16);   // K+16..K+23
    return __builtin_bit_cast(v16bf, t);
}

__device__ __forceinline__ v8f wmma_bf16(v16bf a, v16bf b, v8f c) {
    return __builtin_amdgcn_wmma_f32_16x16x32_bf16(false, a, false, b, (short)0, c, false, false);
}

// ---------------------------------------------------------------------------
// Kernel 1: gate_up GEMM + SwiGLU, output bf16 act[e][t][f] in workspace.
// Block tile: 128 rows x 64 cols (of both gate and up halves), K-tile 64.
// 8 waves: 4 (M) x 2 (N), each wave owns 32x32 per matrix (2x2 WMMA frags).
// ---------------------------------------------------------------------------
extern "C" __global__ __launch_bounds__(256)
void moe_gate_up_silu(const float* __restrict__ x,
                      const float* __restrict__ w1,
                      __bf16* __restrict__ act) {
    __shared__ __align__(16) __bf16 ldsA[128 * LDA];
    __shared__ __align__(16) __bf16 ldsBg[64 * LDB];
    __shared__ __align__(16) __bf16 ldsBu[64 * LDB];

    const int e  = blockIdx.z;
    const int m0 = blockIdx.y * 128;
    const int tf = blockIdx.x * 64;

    const int tid  = threadIdx.x;
    const int lane = tid & 31;
    const int wave = tid >> 5;
    const int wm   = wave & 3;   // 0..3 -> 32-row slabs
    const int wn   = wave >> 2;  // 0..1 -> 32-col slabs

    const float* Xb = x  + ((size_t)e * NT + m0) * NH;
    const float* Wg = w1 + (size_t)e * NH * N2F + tf;
    const float* Wu = Wg + NF;

    v8f accg[2][2] = {};
    v8f accu[2][2] = {};

    const int ac4 = tid & 15;   // 16 float4 per 64-wide A row
    const int ar  = tid >> 4;   // 0..15
    const int bc4 = tid & 15;   // 16 float4 per 64-wide B row
    const int br  = tid >> 4;   // 0..15

    for (int kt = 0; kt < NH; kt += 64) {
        // ---- stage A tile 128x64 fp32 -> bf16 LDS (row-major) ----
        #pragma unroll
        for (int j = 0; j < 8; ++j) {
            int row = ar + j * 16;
            float4 v = *(const float4*)(Xb + (size_t)row * NH + kt + ac4 * 4);
            *(uint2*)&ldsA[row * LDA + ac4 * 4] = make_uint2(pk2(v.x, v.y), pk2(v.z, v.w));
        }
        // ---- stage gate/up B tiles 64x64 fp32 -> bf16 LDS transposed [N][K] ----
        #pragma unroll
        for (int j = 0; j < 4; ++j) {
            int kr = br + j * 16;
            float4 g = *(const float4*)(Wg + (size_t)(kt + kr) * N2F + bc4 * 4);
            float4 u = *(const float4*)(Wu + (size_t)(kt + kr) * N2F + bc4 * 4);
            int n = bc4 * 4;
            ldsBg[(n + 0) * LDB + kr] = (__bf16)g.x;
            ldsBg[(n + 1) * LDB + kr] = (__bf16)g.y;
            ldsBg[(n + 2) * LDB + kr] = (__bf16)g.z;
            ldsBg[(n + 3) * LDB + kr] = (__bf16)g.w;
            ldsBu[(n + 0) * LDB + kr] = (__bf16)u.x;
            ldsBu[(n + 1) * LDB + kr] = (__bf16)u.y;
            ldsBu[(n + 2) * LDB + kr] = (__bf16)u.z;
            ldsBu[(n + 3) * LDB + kr] = (__bf16)u.w;
        }
        // prefetch next K-tile of A into L2 (global_prefetch_b8)
        if (kt + 64 < NH)
            __builtin_prefetch(Xb + (size_t)ar * NH + kt + 64 + ac4 * 4, 0, 1);
        __syncthreads();

        #pragma unroll
        for (int k0 = 0; k0 < 64; k0 += 32) {
            v16bf a0  = ld_frag_a(ldsA,  wm * 32,      k0, lane);
            v16bf a1  = ld_frag_a(ldsA,  wm * 32 + 16, k0, lane);
            v16bf bg0 = ld_frag_b(ldsBg, wn * 32,      k0, lane);
            v16bf bg1 = ld_frag_b(ldsBg, wn * 32 + 16, k0, lane);
            v16bf bu0 = ld_frag_b(ldsBu, wn * 32,      k0, lane);
            v16bf bu1 = ld_frag_b(ldsBu, wn * 32 + 16, k0, lane);
            accg[0][0] = wmma_bf16(a0, bg0, accg[0][0]);
            accg[0][1] = wmma_bf16(a0, bg1, accg[0][1]);
            accg[1][0] = wmma_bf16(a1, bg0, accg[1][0]);
            accg[1][1] = wmma_bf16(a1, bg1, accg[1][1]);
            accu[0][0] = wmma_bf16(a0, bu0, accu[0][0]);
            accu[0][1] = wmma_bf16(a0, bu1, accu[0][1]);
            accu[1][0] = wmma_bf16(a1, bu0, accu[1][0]);
            accu[1][1] = wmma_bf16(a1, bu1, accu[1][1]);
        }
        __syncthreads();
    }

    // ---- fused SwiGLU epilogue: act = up * silu(gate), bf16 store ----
    const int hi = lane >> 4, ln = lane & 15;
    #pragma unroll
    for (int i = 0; i < 2; ++i) {
        #pragma unroll
        for (int j = 0; j < 2; ++j) {
            int row0 = m0 + wm * 32 + i * 16 + hi * 8;  // + r
            int col  = tf + wn * 32 + j * 16 + ln;      // gate/up shared f index
            #pragma unroll
            for (int r = 0; r < 8; ++r) {
                float g = accg[i][j][r];
                float u = accu[i][j][r];
                float a = u * (g / (1.0f + __expf(-g)));
                act[((size_t)e * NT + row0 + r) * NF + col] = (__bf16)a;
            }
        }
    }
}

// ---------------------------------------------------------------------------
// Kernel 2: down-proj GEMM, bf16 act @ fp32->bf16 W2 -> fp32 out.
// Block tile 128x128, K-tile 64. 8 waves: 4 (M) x 2 (N), wave owns 32x64.
// A tile (already bf16) is staged by the Tensor Data Mover: one
// tensor_load_to_lds per K-tile moves the 128x64x2B tile with LDS padding
// (interval 32 dwords, amount 4 dwords -> stride 72 elements).
// ---------------------------------------------------------------------------
extern "C" __global__ __launch_bounds__(256)
void moe_down(const __bf16* __restrict__ act,
              const float* __restrict__ w2,
              float* __restrict__ out) {
    __shared__ __align__(16) __bf16 ldsA[128 * LDA];
    __shared__ __align__(16) __bf16 ldsB[128 * LDB];

    const int e  = blockIdx.z;
    const int m0 = blockIdx.y * 128;
    const int n0 = blockIdx.x * 128;

    const int tid  = threadIdx.x;
    const int lane = tid & 31;
    const int wave = tid >> 5;
    const int wm   = wave & 3;
    const int wn   = wave >> 2;

    const __bf16* Ab = act + ((size_t)e * NT + m0) * NF;
    const float*  Wb = w2  + (size_t)e * NF * NH + n0;

    v8f acc[2][4] = {};

    const int bc4 = tid & 31;   // 32 float4 per 128-wide B row
    const int br  = tid >> 5;   // 0..7
#if !USE_TDM
    const int ac8 = tid & 7;    // 8 uint4 (8 bf16 each) per 64-wide A row
    const int ar  = tid >> 3;   // 0..31
#endif

#if USE_TDM
    const unsigned ldsA_off = (unsigned)(size_t)(&ldsA[0]);  // AS3 byte offset
#endif

    for (int kt = 0; kt < NF; kt += 64) {
#if USE_TDM
        // ---- TDM: DMA the 128x64 bf16 A tile into padded LDS (wave 0 only;
        //      EXEC is ignored by TENSOR_LOAD_TO_LDS, one op per workgroup) ----
        if (wave == 0) {
            unsigned long long ga = (unsigned long long)(size_t)(Ab + kt);
            u32x4 g0;
            g0.x = 1u;                                    // count=1, user D#
            g0.y = ldsA_off;                              // lds_addr (bytes)
            g0.z = (unsigned)ga;                          // global_addr[31:0]
            g0.w = (unsigned)(ga >> 32) | 0x80000000u;    // global_addr[56:32] | type=2
            i32x8 g1;
            g1[0] = (int)((1u << 16)      // data_size = 2 bytes
                        | (1u << 20)      // pad_enable
                        | (4u << 22)      // pad_interval: 2^(4+1)=32 dwords (=128B row)
                        | (3u << 25));    // pad_amount: 4 dwords (=16B -> stride 72 elems)
            g1[1] = (int)((NF & 0xFFFFu) << 16);          // tensor_dim0[15:0] in w1[31:16]
            g1[2] = (int)(((NF >> 16) & 0xFFFFu)          // tensor_dim0[31:16]
                        | (128u << 16));                  // tensor_dim1[15:0] = 128 rows
            g1[3] = (int)(64u << 16);                     // tensor_dim1 hi=0 | tile_dim0=64
            g1[4] = (int)(128u);                          // tile_dim1=128, tile_dim2=0
            g1[5] = (int)(NF);                            // tensor_dim0_stride lo32 = 5632
            g1[6] = 0;                                    // stride hi, dim1_stride lo
            g1[7] = 0;
            i32x4 z4 = {0, 0, 0, 0};
#if __has_include(<hip/amd_detail/amd_gfx1250_TDM.h>)
            i32x8 z8 = {0, 0, 0, 0, 0, 0, 0, 0};
            __builtin_amdgcn_tensor_load_to_lds(g0, g1, z4, z4, z8, 0);
#else
            __builtin_amdgcn_tensor_load_to_lds(g0, g1, z4, z4, 0);
#endif
            __builtin_amdgcn_s_wait_tensorcnt(0);
        }
#else
        // ---- fallback: per-thread staging of A tile 128x64 (already bf16) ----
        #pragma unroll
        for (int j = 0; j < 4; ++j) {
            int row = ar + j * 32;
            uint4 v = *(const uint4*)(Ab + (size_t)row * NF + kt + ac8 * 8);
            *(uint4*)&ldsA[row * LDA + ac8 * 8] = v;
        }
#endif
        // ---- stage B tile 64x128 fp32 -> bf16 transposed [N][K] ----
        #pragma unroll
        for (int j = 0; j < 8; ++j) {
            int kr = br + j * 8;
            float4 v = *(const float4*)(Wb + (size_t)(kt + kr) * NH + bc4 * 4);
            int n = bc4 * 4;
            ldsB[(n + 0) * LDB + kr] = (__bf16)v.x;
            ldsB[(n + 1) * LDB + kr] = (__bf16)v.y;
            ldsB[(n + 2) * LDB + kr] = (__bf16)v.z;
            ldsB[(n + 3) * LDB + kr] = (__bf16)v.w;
        }
        if (kt + 64 < NF)
            __builtin_prefetch(Wb + (size_t)(kt + 64 + br) * NH + bc4 * 4, 0, 1);
        __syncthreads();

        #pragma unroll
        for (int k0 = 0; k0 < 64; k0 += 32) {
            v16bf a0 = ld_frag_a(ldsA, wm * 32,      k0, lane);
            v16bf a1 = ld_frag_a(ldsA, wm * 32 + 16, k0, lane);
            #pragma unroll
            for (int jn = 0; jn < 4; ++jn) {
                v16bf b = ld_frag_b(ldsB, wn * 64 + jn * 16, k0, lane);
                acc[0][jn] = wmma_bf16(a0, b, acc[0][jn]);
                acc[1][jn] = wmma_bf16(a1, b, acc[1][jn]);
            }
        }
        __syncthreads();
    }

    const int hi = lane >> 4, ln = lane & 15;
    #pragma unroll
    for (int i = 0; i < 2; ++i) {
        #pragma unroll
        for (int j = 0; j < 4; ++j) {
            int row0 = m0 + wm * 32 + i * 16 + hi * 8;
            int col  = n0 + wn * 64 + j * 16 + ln;
            #pragma unroll
            for (int r = 0; r < 8; ++r)
                out[((size_t)e * NT + row0 + r) * NH + col] = acc[i][j][r];
        }
    }
}

// ---------------------------------------------------------------------------
extern "C" void kernel_launch(void* const* d_in, const int* in_sizes, int n_in,
                              void* d_out, int out_size, void* d_ws, size_t ws_size,
                              hipStream_t stream) {
    const float* x  = (const float*)d_in[0];   // [E*T, H] fp32
    const float* w1 = (const float*)d_in[1];   // [E, H, 2F] fp32
    const float* w2 = (const float*)d_in[2];   // [E, F, H] fp32
    float* out = (float*)d_out;                // [E*T, H] fp32
    __bf16* actb = (__bf16*)d_ws;              // [E, T, F] bf16 (92 MB)

    dim3 blk(256);
    dim3 g1(NF / 64, NT / 128, NE);   // 88 x 8 x 8
    hipLaunchKernelGGL(moe_gate_up_silu, g1, blk, 0, stream, x, w1, actb);

    dim3 g2(NH / 128, NT / 128, NE);  // 16 x 8 x 8
    hipLaunchKernelGGL(moe_down, g2, blk, 0, stream, actb, w2, out);
}
